// DeepSeekPretrainedMoE_24919400252143
// MI455X (gfx1250) — matrix operand order repaired and checked by reading, not compiled, not run on hardware
//
#include <hip/hip_runtime.h>
#include <hip/hip_bf16.h>

// ---------------------------------------------------------------------------
// DeepSeek-style decoder layer for MI455X (gfx1250, wave32, WMMA bf16)
// - All GEMMs + attention on v_wmma_f32_16x16x32_bf16 (fp32 accumulate)
// - Double-buffered LDS ping-pong GEMM main loop
// - A-operand tiles DMA'd into LDS by the Tensor Data Mover (TENSORcnt)
// - fp32 weights converted to bf16 in-register while staging into LDS
// ---------------------------------------------------------------------------

#define USE_TDM 1

typedef unsigned short u16;
typedef unsigned int   u32;
typedef __attribute__((ext_vector_type(16))) __bf16 bf16x16;
typedef __attribute__((ext_vector_type(8)))  float  f32x8;
typedef __attribute__((ext_vector_type(4)))  float  f32x4;
typedef __attribute__((ext_vector_type(4)))  u32    u32x4;
typedef __attribute__((ext_vector_type(4)))  int    i32x4;
typedef __attribute__((ext_vector_type(8)))  int    i32x8;

union Frag { u32x4 u[2]; bf16x16 v; };

static constexpr int TOK    = 2048;   // B*S
static constexpr int DMODEL = 4096;
static constexpr int NHEAD  = 32;
static constexpr int DHEAD  = 128;
static constexpr int FFDIM  = 11008;
static constexpr int SEQ    = 1024;
static constexpr int DIN    = 512;
static constexpr int DOUT   = 512;

__device__ __forceinline__ u16 f2bf(float f) {
  u32 u = __float_as_uint(f);
  u32 r = u + 0x7FFFu + ((u >> 16) & 1u);   // round-to-nearest-even
  return (u16)(r >> 16);
}

__device__ __forceinline__ f32x8 zero8() {
  f32x8 z;
#pragma unroll
  for (int i = 0; i < 8; ++i) z[i] = 0.0f;
  return z;
}

#define WMMA_BF16(a, b, c) \
  __builtin_amdgcn_wmma_f32_16x16x32_bf16(false, (a), false, (b), (short)0, (c), false, false)

#if USE_TDM
// ---------------------------------------------------------------------------
// Tensor Data Mover: DMA a 128x32 bf16 tile (row stride = row_elems) into LDS.
// D# built per CDNA5 ISA:
//  g0: count=1 | lds_addr | global_addr[56:0] | type=2
//  g1: data_size=2B, tensor_dim0=row_elems, tensor_dim1=2^20,
//      tile_dim0=32, tile_dim1=128, tensor_dim0_stride=row_elems
// Issued by one wave only; completion tracked on TENSORcnt.
// ---------------------------------------------------------------------------
__device__ __forceinline__ void tdm_load_tile_bf16(const u16* gsrc, u32 lds_byte_off,
                                                   u32 row_elems) {
  unsigned long long ga = (unsigned long long)(size_t)gsrc;
  u32x4 g0;
  g0[0] = 1u;                                  // count=1 (valid), user mode
  g0[1] = lds_byte_off;                        // LDS byte address
  g0[2] = (u32)ga;                             // global_addr[31:0]
  g0[3] = (u32)(ga >> 32) | 0x80000000u;       // global_addr[56:32] | type=2
  i32x8 g1;
  g1[0] = (int)(1u << 16);                     // workgroup_mask=0, data_size=1 (2 bytes)
  g1[1] = (int)((row_elems & 0xffffu) << 16);  // tensor_dim0[15:0]
  g1[2] = (int)(row_elems >> 16);              // tensor_dim0[31:16] | tensor_dim1[15:0]=0
  g1[3] = (int)((32u << 16) | 16u);            // tensor_dim1[31:16]=16 (2^20) | tile_dim0=32
  g1[4] = 128;                                 // tile_dim1=128, tile_dim2=0
  g1[5] = (int)row_elems;                      // tensor_dim0_stride[31:0]
  g1[6] = 0;                                   // stride hi | dim1_stride lo
  g1[7] = 0;
  i32x4 z4 = {0, 0, 0, 0};
#if __clang_major__ >= 23
  i32x8 z8 = {0, 0, 0, 0, 0, 0, 0, 0};
  __builtin_amdgcn_tensor_load_to_lds(g0, g1, z4, z4, z8, 0);
#else
  __builtin_amdgcn_tensor_load_to_lds(g0, g1, z4, z4, 0);
#endif
}
#endif

// ---------------------------------------------------------------------------
// fp32 -> bf16 elementwise convert
// ---------------------------------------------------------------------------
__global__ __launch_bounds__(256)
void cvt_bf16_k(const float* __restrict__ in, u16* __restrict__ out, int n) {
  int i = blockIdx.x * 256 + threadIdx.x;
  if (i < n) out[i] = f2bf(in[i]);
}

// ---------------------------------------------------------------------------
// Tiled WMMA GEMM:  C[M,N] = A_bf16[M,K] @ B_f32[K,N]  (+ epilogue)
// Block tile 128x128, BK=32, 256 threads = 8 waves, wave tile 64x32.
// Double-buffered: TDM streams A tiles to LDS, weight loads held in VGPRs
// across the compute phase, one barrier per K-step.
// ---------------------------------------------------------------------------
enum { EPI_F32 = 0, EPI_F32_BIAS = 1, EPI_BF16 = 2, EPI_ADD_F32 = 3, EPI_SILU_MUL_BF16 = 4 };

template <int EPI>
__global__ __launch_bounds__(256)
void gemm_bf16_k(const u16*   __restrict__ A,
                 const float* __restrict__ B,
                 int M, int N, int K,
                 const float* __restrict__ bias,
                 const float* __restrict__ aux,
                 float* __restrict__ Cf,
                 u16*   __restrict__ Cb) {
  __shared__ __align__(16) u16 As[2 * 128 * 32];   // [buf][m][k]
  __shared__ __align__(16) u16 Bs[2 * 128 * 32];   // [buf][n][k] bf16 weights

  const int tid  = threadIdx.x;
  const int lane = tid & 31;
  const int wid  = tid >> 5;
  const int wm   = wid >> 2;        // 0..1
  const int wn   = wid & 3;         // 0..3
  const int half = lane >> 4;       // 0/1
  const int r    = lane & 15;
  const int m0   = blockIdx.y * 128;
  const int n0   = blockIdx.x * 128;
  const int nk   = K / 32;

  // per-thread B staging coordinates (32x128 f32 panel, 4 float4 per thread)
  int bkk[4], bnb[4];
#pragma unroll
  for (int c = 0; c < 4; ++c) {
    int chunk = tid + c * 256;       // 0..1023
    bkk[c] = chunk >> 5;             // 0..31
    bnb[c] = (chunk & 31) * 4;       // 0..124
  }

  f32x8 acc[4][2];
#pragma unroll
  for (int fr = 0; fr < 4; ++fr)
#pragma unroll
    for (int fc = 0; fc < 2; ++fc) acc[fr][fc] = zero8();

  // ---------------- prologue: stage tile 0 into buffer 0 ----------------
#if USE_TDM
  if (wid == 0)
    tdm_load_tile_bf16(A + (size_t)m0 * K, (u32)(size_t)As, (u32)K);
#else
#pragma unroll
  for (int c = 0; c < 2; ++c) {
    int chunk = tid + c * 256;
    int row = chunk >> 2, colc = (chunk & 3) * 8;
    ((u32x4*)As)[chunk] = *(const u32x4*)(A + (size_t)(m0 + row) * K + colc);
  }
#endif
  f32x4 breg[4];
#pragma unroll
  for (int c = 0; c < 4; ++c)
    breg[c] = *(const f32x4*)(B + (size_t)bkk[c] * N + n0 + bnb[c]);
#pragma unroll
  for (int c = 0; c < 4; ++c) {
#pragma unroll
    for (int j = 0; j < 4; ++j)
      Bs[(bnb[c] + j) * 32 + bkk[c]] = f2bf(breg[c][j]);
  }
#if USE_TDM
  if (wid == 0) __builtin_amdgcn_s_wait_tensorcnt(0);
#endif
  __syncthreads();

  // ---------------- main loop: ping-pong ----------------
  for (int kt = 0; kt < nk; ++kt) {
    const int cur = kt & 1, nxt = cur ^ 1;
    const bool more = (kt + 1 < nk);
    const u16* as = As + cur * 4096;
    const u16* bs = Bs + cur * 4096;

    if (more) {
      // issue next A tile via TDM and next B panel into registers (in flight
      // during the WMMA phase below)
#if USE_TDM
      if (wid == 0)
        tdm_load_tile_bf16(A + (size_t)m0 * K + (kt + 1) * 32,
                           (u32)(size_t)(As + nxt * 4096), (u32)K);
#endif
#pragma unroll
      for (int c = 0; c < 4; ++c)
        breg[c] = *(const f32x4*)(B + (size_t)((kt + 1) * 32 + bkk[c]) * N + n0 + bnb[c]);
    }

    // ---- compute current tile: 8 wmma per wave ----
    Frag a[4], b[2];
#pragma unroll
    for (int fr = 0; fr < 4; ++fr) {
      const u32x4* p = (const u32x4*)(as + (wm * 64 + fr * 16 + r) * 32 + half * 16);
      a[fr].u[0] = p[0];          // K 16h .. 16h+7
      a[fr].u[1] = p[1];          // K 16h+8 .. 16h+15
    }
#pragma unroll
    for (int fc = 0; fc < 2; ++fc) {
      const u32x4* p = (const u32x4*)(bs + (wn * 32 + fc * 16 + r) * 32 + half * 16);
      b[fc].u[0] = p[0];          // K 16h .. 16h+15 (contiguous)
      b[fc].u[1] = p[1];
    }
#pragma unroll
    for (int fr = 0; fr < 4; ++fr)
#pragma unroll
      for (int fc = 0; fc < 2; ++fc)
        acc[fr][fc] = WMMA_BF16(a[fr].v, b[fc].v, acc[fr][fc]);

    if (more) {
      // commit next B panel (bf16, transposed) into the other buffer
      u16* bsn = Bs + nxt * 4096;
#pragma unroll
      for (int c = 0; c < 4; ++c) {
#pragma unroll
        for (int j = 0; j < 4; ++j)
          bsn[(bnb[c] + j) * 32 + bkk[c]] = f2bf(breg[c][j]);
      }
#if USE_TDM
      if (wid == 0) __builtin_amdgcn_s_wait_tensorcnt(0);
#else
#pragma unroll
      for (int c = 0; c < 2; ++c) {
        int chunk = tid + c * 256;
        int row = chunk >> 2, colc = (chunk & 3) * 8;
        ((u32x4*)(As + nxt * 4096))[chunk] =
            *(const u32x4*)(A + (size_t)(m0 + row) * K + (kt + 1) * 32 + colc);
      }
#endif
    }
    __syncthreads();
  }

  // ---------------- epilogue ----------------
#pragma unroll
  for (int fr = 0; fr < 4; ++fr) {
    int mrow = m0 + wm * 64 + fr * 16 + half * 8;
#pragma unroll
    for (int fc = 0; fc < 2; ++fc) {
      int col = n0 + wn * 32 + fc * 16 + r;
      float bv = (EPI == EPI_F32_BIAS) ? bias[col] : 0.0f;
#pragma unroll
      for (int v = 0; v < 8; ++v) {
        size_t idx = (size_t)(mrow + v) * N + col;
        float x = acc[fr][fc][v];
        if (EPI == EPI_F32)            Cf[idx] = x;
        else if (EPI == EPI_F32_BIAS)  Cf[idx] = x + bv;
        else if (EPI == EPI_BF16)      Cb[idx] = f2bf(x);
        else if (EPI == EPI_ADD_F32)   Cf[idx] += x;
        else {                          // fused silu(gate)*up -> bf16
          float s = x / (1.0f + __expf(-x));
          Cb[idx] = f2bf(s * aux[idx]);
        }
      }
    }
  }
}

// ---------------------------------------------------------------------------
// RMSNorm: fp32 in -> bf16 out (one workgroup per token)
// ---------------------------------------------------------------------------
__global__ __launch_bounds__(256)
void rmsnorm_k(const float* __restrict__ h, const float* __restrict__ w,
               u16* __restrict__ out, int D) {
  __shared__ float red[256];
  const int t = blockIdx.x;
  const float* hp = h + (size_t)t * D;
  float s = 0.f;
  for (int i = threadIdx.x; i < D; i += 256) { float v = hp[i]; s += v * v; }
  red[threadIdx.x] = s;
  __syncthreads();
  for (int off = 128; off > 0; off >>= 1) {
    if (threadIdx.x < off) red[threadIdx.x] += red[threadIdx.x + off];
    __syncthreads();
  }
  float rms = rsqrtf(red[0] / (float)D + 1e-6f);
  for (int i = threadIdx.x; i < D; i += 256)
    out[(size_t)t * D + i] = f2bf(hp[i] * rms * w[i]);
}

// ---------------------------------------------------------------------------
// Flash attention (causal): one workgroup (4 waves) per (qblock64, head, batch)
// QK^T and P*V on WMMA bf16; online softmax in fp32.
// ---------------------------------------------------------------------------
__global__ __launch_bounds__(128)
void attn_k(const u16* __restrict__ Q, const u16* __restrict__ Kc,
            const u16* __restrict__ V, u16* __restrict__ O) {
  __shared__ __align__(16) u16 Qs[64 * 128];   // [q][dh]
  __shared__ __align__(16) u16 Ks[64 * 128];   // [kv][dh]  (== Bt for QK^T)
  __shared__ __align__(16) u16 Vt[128 * 64];   // [dh][kv]  (== Bt for P*V)
  __shared__ __align__(16) u16 Ps[4][16 * 64]; // per-wave P tile [q16][kv64]

  const int tid  = threadIdx.x;
  const int lane = tid & 31;
  const int w    = tid >> 5;        // wave 0..3 owns q rows w*16..+15
  const int half = lane >> 4;
  const int r    = lane & 15;
  const int qb   = blockIdx.x;      // q block (64 rows)
  const int hh   = blockIdx.y;      // head
  const int bb   = blockIdx.z;      // batch
  const int cb   = hh * DHEAD;      // channel base
  const int tok0 = bb * SEQ;        // batch token base
  const float SCALE = 0.08838834764831845f;   // 1/sqrt(128)

  // stage Q tile once
#pragma unroll
  for (int i = 0; i < 8; ++i) {
    int c = tid + i * 128;                     // 0..1023 16B chunks
    int row = c >> 4, colc = (c & 15) * 8;
    ((u32x4*)Qs)[c] =
        *(const u32x4*)(Q + (size_t)(tok0 + qb * 64 + row) * DMODEL + cb + colc);
  }

  f32x8 oacc[8];
#pragma unroll
  for (int nt = 0; nt < 8; ++nt) oacc[nt] = zero8();
  float m_i[8], l_i[8];
#pragma unroll
  for (int v = 0; v < 8; ++v) { m_i[v] = -1e30f; l_i[v] = 0.0f; }

  for (int j = 0; j <= qb; ++j) {
    __syncthreads();
    // stage K tile [64][128]
#pragma unroll
    for (int i = 0; i < 8; ++i) {
      int c = tid + i * 128;
      int row = c >> 4, colc = (c & 15) * 8;
      ((u32x4*)Ks)[c] =
          *(const u32x4*)(Kc + (size_t)(tok0 + j * 64 + row) * DMODEL + cb + colc);
    }
    // stage V tile transposed -> Vt[dh][kv]
#pragma unroll
    for (int i = 0; i < 32; ++i) {
      int c  = tid + i * 128;                  // 0..4095 u32 chunks
      int kv = c >> 6;
      int d2 = c & 63;
      u32 uu = *(const u32*)(V + (size_t)(tok0 + j * 64 + kv) * DMODEL + cb + d2 * 2);
      Vt[(d2 * 2 + 0) * 64 + kv] = (u16)(uu & 0xffffu);
      Vt[(d2 * 2 + 1) * 64 + kv] = (u16)(uu >> 16);
    }
    __syncthreads();

    // ---- S = Q K^T (16 rows x 64 cols per wave) ----
    f32x8 sacc[4];
#pragma unroll
    for (int nf = 0; nf < 4; ++nf) sacc[nf] = zero8();
#pragma unroll
    for (int ks = 0; ks < 4; ++ks) {
      Frag aq;
      {
        const u32x4* p = (const u32x4*)(Qs + (w * 16 + r) * 128 + ks * 32 + half * 16);
        aq.u[0] = p[0]; aq.u[1] = p[1];
      }
#pragma unroll
      for (int nf = 0; nf < 4; ++nf) {
        Frag bk;
        const u32x4* p = (const u32x4*)(Ks + (nf * 16 + r) * 128 + ks * 32 + half * 16);
        bk.u[0] = p[0]; bk.u[1] = p[1];
        sacc[nf] = WMMA_BF16(aq.v, bk.v, sacc[nf]);
      }
    }

    // ---- online softmax update ----
    float mx[8];
#pragma unroll
    for (int v = 0; v < 8; ++v) {
      float m_ = -1e30f;
#pragma unroll
      for (int nf = 0; nf < 4; ++nf) {
        float s = sacc[nf][v] * SCALE;
        if (j == qb) {
          int rowg = w * 16 + v + half * 8;
          int colg = nf * 16 + r;
          if (colg > rowg) s = -1e30f;         // causal mask on diagonal block
        }
        sacc[nf][v] = s;
        m_ = fmaxf(m_, s);
      }
#pragma unroll
      for (int msk = 8; msk >= 1; msk >>= 1)
        m_ = fmaxf(m_, __shfl_xor(m_, msk, 32));
      mx[v] = m_;
    }

#pragma unroll
    for (int v = 0; v < 8; ++v) {
      float mnew  = fmaxf(m_i[v], mx[v]);
      float alpha = __expf(m_i[v] - mnew);
      float rs = 0.0f;
#pragma unroll
      for (int nf = 0; nf < 4; ++nf) {
        float p = __expf(sacc[nf][v] - mnew);
        sacc[nf][v] = p;
        rs += p;
      }
#pragma unroll
      for (int msk = 8; msk >= 1; msk >>= 1) rs += __shfl_xor(rs, msk, 32);
      l_i[v] = l_i[v] * alpha + rs;
      m_i[v] = mnew;
#pragma unroll
      for (int nt = 0; nt < 8; ++nt) oacc[nt][v] = oacc[nt][v] * alpha;
      // scatter P (C-layout) into per-wave LDS tile
#pragma unroll
      for (int nf = 0; nf < 4; ++nf)
        Ps[w][(v + half * 8) * 64 + nf * 16 + r] = f2bf(sacc[nf][v]);
    }
    // per-wave LDS RAW fence before re-reading Ps in A-fragment layout
    asm volatile("s_wait_dscnt 0" ::: "memory");

    // ---- O += P V ----
#pragma unroll
    for (int ks2 = 0; ks2 < 2; ++ks2) {
      Frag ap;
      {
        const u32x4* p = (const u32x4*)(&Ps[w][0] + r * 64 + ks2 * 32 + half * 16);
        ap.u[0] = p[0]; ap.u[1] = p[1];
      }
#pragma unroll
      for (int nt = 0; nt < 8; ++nt) {
        Frag bv;
        const u32x4* p = (const u32x4*)(Vt + (nt * 16 + r) * 64 + ks2 * 32 + half * 16);
        bv.u[0] = p[0]; bv.u[1] = p[1];
        oacc[nt] = WMMA_BF16(ap.v, bv.v, oacc[nt]);
      }
    }
  }

  // ---- finalize: O / l, store bf16 ----
#pragma unroll
  for (int v = 0; v < 8; ++v) {
    float inv = 1.0f / l_i[v];
    size_t rowtok = (size_t)(tok0 + qb * 64 + w * 16 + v + half * 8);
#pragma unroll
    for (int nt = 0; nt < 8; ++nt)
      O[rowtok * DMODEL + cb + nt * 16 + r] = f2bf(oacc[nt][v] * inv);
  }
}

// ---------------------------------------------------------------------------
// Router + gather-mean.  top_k(softmax(x)) indices == top_k(x) indices.
// agg[t] = 0.5*(h[t][i0] + h[t][i1]).  One wave per token.
// ---------------------------------------------------------------------------
__global__ __launch_bounds__(256)
void router_k(const float* __restrict__ h, const float* __restrict__ Wr,
              const float* __restrict__ br, float* __restrict__ agg, int T) {
  const int lane = threadIdx.x & 31;
  const int wid  = threadIdx.x >> 5;
  const int t    = blockIdx.x * 8 + wid;
  if (t >= T) return;
  const float* hp = h + (size_t)t * DMODEL;
  float logit[8];
#pragma unroll
  for (int e = 0; e < 8; ++e) {
    float s = 0.f;
    for (int i = lane; i < DMODEL; i += 32) s += hp[i] * Wr[(size_t)i * 8 + e];
#pragma unroll
    for (int m = 16; m >= 1; m >>= 1) s += __shfl_xor(s, m, 32);
    logit[e] = s + br[e];
  }
  int i0 = 0;
#pragma unroll
  for (int e = 1; e < 8; ++e) if (logit[e] > logit[i0]) i0 = e;
  int i1 = (i0 == 0) ? 1 : 0;
#pragma unroll
  for (int e = 0; e < 8; ++e) if (e != i0 && logit[e] > logit[i1]) i1 = e;
  if (lane == 0) agg[t] = 0.5f * (hp[i0] + hp[i1]);
}

// ---------------------------------------------------------------------------
// colsum(W_out) and rank-1 final projection:
// out[t,j] = agg[t] * colsum[j] + b_out[j]
// ---------------------------------------------------------------------------
__global__ __launch_bounds__(256)
void colsum_k(const float* __restrict__ W, float* __restrict__ cs, int Kd, int Nd) {
  int j = blockIdx.x * 256 + threadIdx.x;
  if (j >= Nd) return;
  float s = 0.f;
  for (int d = 0; d < Kd; ++d) s += W[(size_t)d * Nd + j];
  cs[j] = s;
}

__global__ __launch_bounds__(256)
void finalout_k(const float* __restrict__ agg, const float* __restrict__ cs,
                const float* __restrict__ bout, float* __restrict__ out,
                int T, int Nd) {
  int idx = blockIdx.x * 256 + threadIdx.x;
  if (idx >= T * Nd) return;
  int t = idx / Nd, j = idx - t * Nd;
  out[idx] = agg[t] * cs[j] + bout[j];
}

// ---------------------------------------------------------------------------
// Host launcher
// ---------------------------------------------------------------------------
extern "C" void kernel_launch(void* const* d_in, const int* in_sizes, int n_in,
                              void* d_out, int out_size, void* d_ws, size_t ws_size,
                              hipStream_t stream) {
  (void)in_sizes; (void)n_in; (void)out_size; (void)ws_size;
  const float* x      = (const float*)d_in[0];
  const float* W_in   = (const float*)d_in[1];
  const float* b_in   = (const float*)d_in[2];
  const float* ln1_w  = (const float*)d_in[3];
  const float* Wq     = (const float*)d_in[4];
  const float* Wk     = (const float*)d_in[5];
  const float* Wv     = (const float*)d_in[6];
  const float* Wo     = (const float*)d_in[7];
  const float* ln2_w  = (const float*)d_in[8];
  const float* W_gate = (const float*)d_in[9];
  const float* W_up   = (const float*)d_in[10];
  const float* W_down = (const float*)d_in[11];
  const float* W_rout = (const float*)d_in[12];
  const float* b_rout = (const float*)d_in[13];
  const float* W_out  = (const float*)d_in[14];
  const float* b_out  = (const float*)d_in[15];
  float* out = (float*)d_out;

  char* wsp = (char*)d_ws;
  auto take = [&](size_t bytes) {
    char* p = wsp;
    wsp += (bytes + 255) & ~(size_t)255;
    return p;
  };
  u16*   x_bf  = (u16*)  take((size_t)TOK * DIN * 2);
  float* h     = (float*)take((size_t)TOK * DMODEL * 4);
  u16*   hn_bf = (u16*)  take((size_t)TOK * DMODEL * 2);
  u16*   q_bf  = (u16*)  take((size_t)TOK * DMODEL * 2);
  u16*   k_bf  = (u16*)  take((size_t)TOK * DMODEL * 2);
  u16*   v_bf  = (u16*)  take((size_t)TOK * DMODEL * 2);
  u16*   c_bf  = (u16*)  take((size_t)TOK * DMODEL * 2);
  float* up_f  = (float*)take((size_t)TOK * FFDIM * 4);
  u16*   ffa   = (u16*)  take((size_t)TOK * FFDIM * 2);
  float* agg   = (float*)take((size_t)TOK * 4);
  float* csum  = (float*)take((size_t)DOUT * 4);

  const dim3 blk256(256), blk128(128);
  const dim3 gD(DMODEL / 128, TOK / 128);   // N=4096 GEMMs
  const dim3 gF(FFDIM / 128, TOK / 128);    // N=11008 GEMMs

  // 1. x -> bf16
  cvt_bf16_k<<<(TOK * DIN) / 256, blk256, 0, stream>>>(x, x_bf, TOK * DIN);
  // 2. h = x @ W_in + b_in
  gemm_bf16_k<EPI_F32_BIAS><<<gD, blk256, 0, stream>>>(
      x_bf, W_in, TOK, DMODEL, DIN, b_in, nullptr, h, nullptr);
  // 3. hn = rmsnorm(h, ln1_w)
  rmsnorm_k<<<TOK, blk256, 0, stream>>>(h, ln1_w, hn_bf, DMODEL);
  // 4. q,k,v projections
  gemm_bf16_k<EPI_BF16><<<gD, blk256, 0, stream>>>(
      hn_bf, Wq, TOK, DMODEL, DMODEL, nullptr, nullptr, nullptr, q_bf);
  gemm_bf16_k<EPI_BF16><<<gD, blk256, 0, stream>>>(
      hn_bf, Wk, TOK, DMODEL, DMODEL, nullptr, nullptr, nullptr, k_bf);
  gemm_bf16_k<EPI_BF16><<<gD, blk256, 0, stream>>>(
      hn_bf, Wv, TOK, DMODEL, DMODEL, nullptr, nullptr, nullptr, v_bf);
  // 5. causal flash attention
  attn_k<<<dim3(SEQ / 64, NHEAD, 2), blk128, 0, stream>>>(q_bf, k_bf, v_bf, c_bf);
  // 6. h += ctx @ Wo
  gemm_bf16_k<EPI_ADD_F32><<<gD, blk256, 0, stream>>>(
      c_bf, Wo, TOK, DMODEL, DMODEL, nullptr, nullptr, h, nullptr);
  // 7. hn = rmsnorm(h, ln2_w)
  rmsnorm_k<<<TOK, blk256, 0, stream>>>(h, ln2_w, hn_bf, DMODEL);
  // 8. up = hn @ W_up
  gemm_bf16_k<EPI_F32><<<gF, blk256, 0, stream>>>(
      hn_bf, W_up, TOK, FFDIM, DMODEL, nullptr, nullptr, up_f, nullptr);
  // 9. ffa = bf16( silu(hn @ W_gate) * up )
  gemm_bf16_k<EPI_SILU_MUL_BF16><<<gF, blk256, 0, stream>>>(
      hn_bf, W_gate, TOK, FFDIM, DMODEL, nullptr, up_f, nullptr, ffa);
  // 10. h += ffa @ W_down
  gemm_bf16_k<EPI_ADD_F32><<<gD, blk256, 0, stream>>>(
      ffa, W_down, TOK, DMODEL, FFDIM, nullptr, nullptr, h, nullptr);
  // 11. router top-2 gather-mean -> agg[t]
  router_k<<<TOK / 8, blk256, 0, stream>>>(h, W_rout, b_rout, agg, TOK);
  // 12-13. out[t,j] = agg[t]*colsum(W_out)[j] + b_out[j]
  colsum_k<<<(DOUT + 255) / 256, blk256, 0, stream>>>(W_out, csum, DMODEL, DOUT);
  finalout_k<<<(TOK * DOUT) / 256, blk256, 0, stream>>>(agg, csum, b_out, out, TOK, DOUT);
}
